// MultiheadAttentionWithCaching_16174846836811
// MI455X (gfx1250) — compile-verified, matrix operand order checked by reading, not run on hardware
//
#include <hip/hip_runtime.h>
#include <hip/hip_bf16.h>
#include <math.h>

// ---------------------------------------------------------------------------
// Causal multi-head attention forward for MI455X (gfx1250, wave32, WMMA).
// Pipeline: cast->bf16, QKV projections (bf16 WMMA), QK^T (masked, scaled),
// fused register-resident row softmax, PV, output projection.
// Tiles are staged via GLOBAL_LOAD_ASYNC_TO_LDS_B128 into a double-buffered
// LDS pipeline (ASYNCcnt + s_wait_asynccnt), feeding v_wmma_f32_16x16x32_bf16.
// ---------------------------------------------------------------------------

typedef __bf16 bf16_t;
typedef __bf16 v16bf __attribute__((ext_vector_type(16)));
typedef float  v8f   __attribute__((ext_vector_type(8)));
typedef unsigned int u32x4 __attribute__((ext_vector_type(4)));

#define B_   4
#define S_   2048
#define D_   2048
#define H_   16
#define HD_  128
#define BS_  (B_ * S_)
#define SCALE_ 0.08838834764831845f   // 1/sqrt(128)

#define TILE_BYTES 10240u             // 128 * 40 * sizeof(bf16)

union FragU { v16bf v; u32x4 q[2]; };

__device__ __forceinline__ bf16_t f2bf(float f) {
    union { float f; unsigned u; } in; in.f = f;
    unsigned u = in.u;
    unsigned r = (u + 0x7FFFu + ((u >> 16) & 1u)) >> 16;   // RNE
    union { unsigned short s; bf16_t b; } out; out.s = (unsigned short)r;
    return out.b;
}

__device__ __forceinline__ unsigned pkbf(unsigned a, unsigned b) {   // 2xf32 bits -> packed bf16
    unsigned ra = (a + 0x7FFFu + ((a >> 16) & 1u)) >> 16;
    unsigned rb = (b + 0x7FFFu + ((b >> 16) & 1u)) >> 16;
    return (ra & 0xFFFFu) | (rb << 16);
}

__device__ __forceinline__ u32x4 cvt8(u32x4 lo, u32x4 hi) {          // 8 f32 -> 8 bf16
    u32x4 r;
    r[0] = pkbf(lo[0], lo[1]); r[1] = pkbf(lo[2], lo[3]);
    r[2] = pkbf(hi[0], hi[1]); r[3] = pkbf(hi[2], hi[3]);
    return r;
}

// ---- CDNA5 async global->LDS copy (ASYNCcnt) ------------------------------
__device__ __forceinline__ unsigned lds_off(const void* p) {
    return (unsigned)(unsigned long long)p;   // low 32 bits of flat = LDS addr
}
__device__ __forceinline__ void async_b128(unsigned lds, const void* g) {
    asm volatile("global_load_async_to_lds_b128 %0, %1, off"
                 :: "v"(lds), "v"(g) : "memory");
}
#define WAIT_ASYNC(n) asm volatile("s_wait_asynccnt " #n ::: "memory")

// ---- shared WMMA micro-kernel: 8x v_wmma per K-step per wave --------------
#define WMMA_STEP(AS, BS) do {                                              \
    FragU af[2], bq[4];                                                     \
    _Pragma("unroll")                                                       \
    for (int i = 0; i < 2; i++) {                                           \
        int r = waveM + i * 16 + mfrag;                                     \
        af[i].q[0] = *(const u32x4*)&AS[r][koff];                           \
        af[i].q[1] = *(const u32x4*)&AS[r][koff + 16];                      \
    }                                                                       \
    _Pragma("unroll")                                                       \
    for (int j = 0; j < 4; j++) {                                           \
        int r = waveN + j * 16 + mfrag;                                     \
        bq[j].q[0] = *(const u32x4*)&BS[r][koff];                           \
        bq[j].q[1] = *(const u32x4*)&BS[r][koff + 16];                      \
    }                                                                       \
    _Pragma("unroll")                                                       \
    for (int i = 0; i < 2; i++)                                             \
        _Pragma("unroll")                                                   \
        for (int j = 0; j < 4; j++)                                         \
            acc[i][j] = __builtin_amdgcn_wmma_f32_16x16x32_bf16(            \
                false, af[i].v, false, bq[j].v, (short)0, acc[i][j],        \
                false, false);                                              \
} while (0)

// ---------------------------------------------------------------------------
// Vectorized fp32 -> bf16 cast: 8 elements per thread-iter (b128 in/out).
// n8 = element count / 8 (all tensor sizes here are multiples of 8).
// ---------------------------------------------------------------------------
__global__ __launch_bounds__(256)
void cast_f32_to_bf16(const float* __restrict__ src, bf16_t* __restrict__ dst, int n8) {
    int i = blockIdx.x * blockDim.x + threadIdx.x;
    int stride = gridDim.x * blockDim.x;
    const u32x4* s = (const u32x4*)src;
    u32x4* d = (u32x4*)dst;
    for (; i < n8; i += stride) {
        u32x4 lo = s[2 * (size_t)i], hi = s[2 * (size_t)i + 1];
        d[i] = cvt8(lo, hi);
    }
}

// ---------------------------------------------------------------------------
// Generic Y = A(bf16,[M,2048]) @ W(bf16,[2048,2048] row=out-col)^T + bias.
// mode 0/1: store bf16 into [B,H,S,hd] (Q and K)
// mode 2  : store bf16 into [B,H,hd,S] (V transposed for PV GEMM)
// mode 3  : store fp32 into [M,2048]    (final output projection, NT stores)
// ---------------------------------------------------------------------------
__global__ __launch_bounds__(256)
void gemm_xw(const bf16_t* __restrict__ A, const bf16_t* __restrict__ W,
             const float* __restrict__ bias, void* __restrict__ dstv, int mode)
{
    __shared__ __align__(16) bf16_t As[2][128][40];
    __shared__ __align__(16) bf16_t Bs[2][128][40];

    const int tid   = threadIdx.x;
    const int lane  = tid & 31;
    const int wid   = tid >> 5;
    const int waveM = (wid & 3) * 32;
    const int waveN = (wid >> 2) * 64;
    const int m0 = blockIdx.y * 128;
    const int n0 = blockIdx.x * 128;
    const int ldr = tid >> 1;          // cooperative-load row 0..127
    const int seg = (tid & 1) * 16;    // 16-element column segment
    const int mfrag = lane & 15;
    const int koff  = (lane >> 4) << 3;

    const bf16_t* ga = A + (size_t)(m0 + ldr) * D_ + seg;
    const bf16_t* gb = W + (size_t)(n0 + ldr) * D_ + seg;
    const unsigned la0 = lds_off(&As[0][ldr][seg]);
    const unsigned lb0 = lds_off(&Bs[0][ldr][seg]);

    // prefetch tile 0 into buffer 0
    async_b128(la0,      ga);
    async_b128(la0 + 16, ga + 8);
    async_b128(lb0,      gb);
    async_b128(lb0 + 16, gb + 8);

    v8f acc[2][4] = {};
    const int nT = D_ / 32;
    for (int t = 0; t < nT; ++t) {
        const int cur = t & 1;
        if (t + 1 < nT) {
            const unsigned la = la0 + (unsigned)(cur ^ 1) * TILE_BYTES;
            const unsigned lb = lb0 + (unsigned)(cur ^ 1) * TILE_BYTES;
            const bf16_t* ga2 = ga + (size_t)(t + 1) * 32;
            const bf16_t* gb2 = gb + (size_t)(t + 1) * 32;
            async_b128(la,      ga2);
            async_b128(la + 16, ga2 + 8);
            async_b128(lb,      gb2);
            async_b128(lb + 16, gb2 + 8);
            WAIT_ASYNC(4);             // tile t resident (in-order completion)
        } else {
            WAIT_ASYNC(0);
        }
        __syncthreads();
        WMMA_STEP(As[cur], Bs[cur]);
        __syncthreads();               // fragment reads drained before reuse
    }

    const int nf = lane & 15;
    const int mb = (lane >> 4) << 3;
#pragma unroll
    for (int j = 0; j < 4; j++) {
        int col = n0 + waveN + j * 16 + nf;
        float bv = bias[col];
#pragma unroll
        for (int i = 0; i < 2; i++) {
#pragma unroll
            for (int g = 0; g < 8; g++) {
                int row = m0 + waveM + i * 16 + mb + g;
                float v = acc[i][j][g] + bv;
                if (mode == 3) {
                    __builtin_nontemporal_store(v, &((float*)dstv)[(size_t)row * D_ + col]);
                } else {
                    int b = row >> 11, s = row & (S_ - 1);
                    int h = col >> 7,  d = col & (HD_ - 1);
                    bf16_t bb = f2bf(v);
                    if (mode == 2)
                        ((bf16_t*)dstv)[(((size_t)b * H_ + h) * HD_ + d) * S_ + s] = bb;
                    else
                        ((bf16_t*)dstv)[(((size_t)b * H_ + h) * S_ + s) * HD_ + d] = bb;
                }
            }
        }
    }
}

// ---------------------------------------------------------------------------
// scores = Q_h @ K_h^T * scale, causal mask -> probs region (fp32).
// Fully-masked 128x128 tiles skipped (softmax writes zeros there).
// ---------------------------------------------------------------------------
__global__ __launch_bounds__(256)
void gemm_qk(const bf16_t* __restrict__ Q, const bf16_t* __restrict__ K,
             float* __restrict__ probs)
{
    const int bh = blockIdx.z;
    const int m0 = blockIdx.y * 128;
    const int n0 = blockIdx.x * 128;
    if (n0 > m0 + 127) return;   // uniform block-level skip

    __shared__ __align__(16) bf16_t As[2][128][40];
    __shared__ __align__(16) bf16_t Bs[2][128][40];

    const bf16_t* Qh = Q + (size_t)bh * S_ * HD_;
    const bf16_t* Kh = K + (size_t)bh * S_ * HD_;

    const int tid   = threadIdx.x;
    const int lane  = tid & 31;
    const int wid   = tid >> 5;
    const int waveM = (wid & 3) * 32;
    const int waveN = (wid >> 2) * 64;
    const int ldr = tid >> 1;
    const int seg = (tid & 1) * 16;
    const int mfrag = lane & 15;
    const int koff  = (lane >> 4) << 3;

    const bf16_t* ga = Qh + (size_t)(m0 + ldr) * HD_ + seg;
    const bf16_t* gb = Kh + (size_t)(n0 + ldr) * HD_ + seg;
    const unsigned la0 = lds_off(&As[0][ldr][seg]);
    const unsigned lb0 = lds_off(&Bs[0][ldr][seg]);

    async_b128(la0,      ga);
    async_b128(la0 + 16, ga + 8);
    async_b128(lb0,      gb);
    async_b128(lb0 + 16, gb + 8);

    v8f acc[2][4] = {};
    const int nT = HD_ / 32;
    for (int t = 0; t < nT; ++t) {
        const int cur = t & 1;
        if (t + 1 < nT) {
            const unsigned la = la0 + (unsigned)(cur ^ 1) * TILE_BYTES;
            const unsigned lb = lb0 + (unsigned)(cur ^ 1) * TILE_BYTES;
            const bf16_t* ga2 = ga + (size_t)(t + 1) * 32;
            const bf16_t* gb2 = gb + (size_t)(t + 1) * 32;
            async_b128(la,      ga2);
            async_b128(la + 16, ga2 + 8);
            async_b128(lb,      gb2);
            async_b128(lb + 16, gb2 + 8);
            WAIT_ASYNC(4);
        } else {
            WAIT_ASYNC(0);
        }
        __syncthreads();
        WMMA_STEP(As[cur], Bs[cur]);
        __syncthreads();
    }

    const int nf = lane & 15;
    const int mb = (lane >> 4) << 3;
    float* Ph = probs + (size_t)bh * S_ * S_;
#pragma unroll
    for (int j = 0; j < 4; j++) {
        int col = n0 + waveN + j * 16 + nf;
#pragma unroll
        for (int i = 0; i < 2; i++) {
#pragma unroll
            for (int g = 0; g < 8; g++) {
                int row = m0 + waveM + i * 16 + mb + g;
                float v = acc[i][j][g] * SCALE_;
                if (col > row) v = -INFINITY;
                Ph[(size_t)row * S_ + col] = v;
            }
        }
    }
}

// ---------------------------------------------------------------------------
// Register-resident causal row softmax: one block per (b,h,q) row, each
// thread holds 8 strided row elements -> exactly one read + one write of the
// 1.07 GB probs tensor (the dominant HBM term at 23.3 TB/s).
// Masked / unwritten tail lanes never read memory; they contribute exp()=0
// and store 0, matching the reference.
// ---------------------------------------------------------------------------
__global__ __launch_bounds__(256)
void softmax_causal(float* __restrict__ probs)
{
    __shared__ float red[256];
    const int tid = threadIdx.x;
    const size_t row = blockIdx.x;
    const int q = (int)(row & (size_t)(S_ - 1));
    float* p = probs + row * (size_t)S_;
    const int len = q + 1;

    float v[8];
#pragma unroll
    for (int k = 0; k < 8; k++) {
        int i = tid + k * 256;
        v[k] = (i < len) ? p[i] : -INFINITY;
    }

    float m = -INFINITY;
#pragma unroll
    for (int k = 0; k < 8; k++) m = fmaxf(m, v[k]);
    red[tid] = m; __syncthreads();
    for (int o = 128; o > 0; o >>= 1) {
        if (tid < o) red[tid] = fmaxf(red[tid], red[tid + o]);
        __syncthreads();
    }
    m = red[0]; __syncthreads();

    float s = 0.f;
#pragma unroll
    for (int k = 0; k < 8; k++) {
        v[k] = __expf(v[k] - m);       // masked lanes: exp(-inf) = 0
        s += v[k];
    }
    red[tid] = s; __syncthreads();
    for (int o = 128; o > 0; o >>= 1) {
        if (tid < o) red[tid] += red[tid + o];
        __syncthreads();
    }
    const float inv = 1.f / red[0];

#pragma unroll
    for (int k = 0; k < 8; k++) p[tid + k * 256] = v[k] * inv;
}

// ---------------------------------------------------------------------------
// attn = probs_h (fp32, packed to bf16 on load) @ V_h (pre-transposed [hd,S]).
// K-loop truncated at causal boundary. Stores merged-head bf16 [B*S, D].
// A tiles: vector fp32 loads + packed bf16 ds_store_b128; B tiles: async LDS.
// ---------------------------------------------------------------------------
__global__ __launch_bounds__(256)
void gemm_pv(const float* __restrict__ probs, const bf16_t* __restrict__ Vt,
             bf16_t* __restrict__ attnb)
{
    const int bh = blockIdx.z;
    const int m0 = blockIdx.y * 128;

    __shared__ __align__(16) bf16_t As[2][128][40];
    __shared__ __align__(16) bf16_t Bs[2][128][40];

    const float*  Ph = probs + (size_t)bh * S_ * S_;
    const bf16_t* Vh = Vt + (size_t)bh * HD_ * S_;

    const int tid   = threadIdx.x;
    const int lane  = tid & 31;
    const int wid   = tid >> 5;
    const int waveM = (wid & 3) * 32;
    const int waveN = (wid >> 2) * 64;
    const int ldr = tid >> 1;
    const int seg = (tid & 1) * 16;
    const int mfrag = lane & 15;
    const int koff  = (lane >> 4) << 3;

    const u32x4* gaf = (const u32x4*)(Ph + (size_t)(m0 + ldr) * S_ + seg);  // 4 u32x4 per 16 f32
    const bf16_t* gb = Vh + (size_t)ldr * S_ + seg;
    const unsigned lb0 = lds_off(&Bs[0][ldr][seg]);

    // prologue: tile 0
    {
        u32x4 w0 = gaf[0], w1 = gaf[1], w2 = gaf[2], w3 = gaf[3];
        u32x4* d = (u32x4*)&As[0][ldr][seg];
        d[0] = cvt8(w0, w1);
        d[1] = cvt8(w2, w3);
    }
    async_b128(lb0,      gb);
    async_b128(lb0 + 16, gb + 8);

    v8f acc[2][4] = {};
    const int nT = (m0 + 128) / 32;    // causal truncation of K loop
    for (int t = 0; t < nT; ++t) {
        const int cur = t & 1;
        if (t + 1 < nT) {
            const int nxt = cur ^ 1;
            const u32x4* ga2 = gaf + (size_t)(t + 1) * 8;   // +32 floats
            u32x4 w0 = ga2[0], w1 = ga2[1], w2 = ga2[2], w3 = ga2[3];
            u32x4* d = (u32x4*)&As[nxt][ldr][seg];
            d[0] = cvt8(w0, w1);
            d[1] = cvt8(w2, w3);
            const bf16_t* gb2 = gb + (size_t)(t + 1) * 32;
            const unsigned lb = lb0 + (unsigned)nxt * TILE_BYTES;
            async_b128(lb,      gb2);
            async_b128(lb + 16, gb2 + 8);
            WAIT_ASYNC(2);
        } else {
            WAIT_ASYNC(0);
        }
        __syncthreads();
        WMMA_STEP(As[cur], Bs[cur]);
        __syncthreads();
    }

    const int nf = lane & 15;
    const int mb = (lane >> 4) << 3;
    const int b = bh >> 4, h = bh & 15;
#pragma unroll
    for (int j = 0; j < 4; j++) {
        int d = waveN + j * 16 + nf;
#pragma unroll
        for (int i = 0; i < 2; i++) {
#pragma unroll
            for (int g = 0; g < 8; g++) {
                int s = m0 + waveM + i * 16 + mb + g;
                attnb[((size_t)b * S_ + s) * D_ + h * HD_ + d] = f2bf(acc[i][j][g]);
            }
        }
    }
}

// ---------------------------------------------------------------------------
extern "C" void kernel_launch(void* const* d_in, const int* in_sizes, int n_in,
                              void* d_out, int out_size, void* d_ws, size_t ws_size,
                              hipStream_t stream)
{
    (void)in_sizes; (void)n_in; (void)out_size; (void)ws_size;

    const float* x  = (const float*)d_in[0];
    const float* Wq = (const float*)d_in[1];
    const float* bq = (const float*)d_in[2];
    const float* Wk = (const float*)d_in[3];
    const float* bk = (const float*)d_in[4];
    const float* Wv = (const float*)d_in[5];
    const float* bv = (const float*)d_in[6];
    const float* Wo = (const float*)d_in[7];
    const float* bo = (const float*)d_in[8];

    float* out   = (float*)d_out;
    float* probs = out + (size_t)BS_ * D_;

    char* ws = (char*)d_ws;
    bf16_t* Xb  = (bf16_t*)ws;  ws += (size_t)BS_ * D_ * 2;
    bf16_t* Wqb = (bf16_t*)ws;  ws += (size_t)D_ * D_ * 2;
    bf16_t* Wkb = (bf16_t*)ws;  ws += (size_t)D_ * D_ * 2;
    bf16_t* Wvb = (bf16_t*)ws;  ws += (size_t)D_ * D_ * 2;
    bf16_t* Wob = (bf16_t*)ws;  ws += (size_t)D_ * D_ * 2;
    bf16_t* Qb  = (bf16_t*)ws;  ws += (size_t)BS_ * D_ * 2;
    bf16_t* Kb  = (bf16_t*)ws;  ws += (size_t)BS_ * D_ * 2;
    bf16_t* Vtb = (bf16_t*)ws;  ws += (size_t)BS_ * D_ * 2;
    bf16_t* Atb = (bf16_t*)ws;  ws += (size_t)BS_ * D_ * 2;

    cast_f32_to_bf16<<<2048, 256, 0, stream>>>(x,  Xb,  BS_ * D_ / 8);
    cast_f32_to_bf16<<<1024, 256, 0, stream>>>(Wq, Wqb, D_ * D_ / 8);
    cast_f32_to_bf16<<<1024, 256, 0, stream>>>(Wk, Wkb, D_ * D_ / 8);
    cast_f32_to_bf16<<<1024, 256, 0, stream>>>(Wv, Wvb, D_ * D_ / 8);
    cast_f32_to_bf16<<<1024, 256, 0, stream>>>(Wo, Wob, D_ * D_ / 8);

    dim3 gProj(D_ / 128, BS_ / 128);                 // 16 x 64
    gemm_xw<<<gProj, 256, 0, stream>>>(Xb, Wqb, bq, Qb,  0);
    gemm_xw<<<gProj, 256, 0, stream>>>(Xb, Wkb, bk, Kb,  1);
    gemm_xw<<<gProj, 256, 0, stream>>>(Xb, Wvb, bv, Vtb, 2);

    dim3 gQK(S_ / 128, S_ / 128, B_ * H_);           // 16 x 16 x 64
    gemm_qk<<<gQK, 256, 0, stream>>>(Qb, Kb, probs);

    softmax_causal<<<B_ * H_ * S_, 256, 0, stream>>>(probs);

    dim3 gPV(1, S_ / 128, B_ * H_);                  // 1 x 16 x 64
    gemm_pv<<<gPV, 256, 0, stream>>>(probs, Vtb, Atb);

    gemm_xw<<<gProj, 256, 0, stream>>>(Atb, Wob, bo, out, 3);
}
